// EdgeWaveAttention_16071767622371
// MI455X (gfx1250) — matrix operand order repaired and checked by reading, not compiled, not run on hardware
//
#include <hip/hip_runtime.h>
#include <hip/hip_bf16.h>

#define BB 8
#define NN 512
#define DD 256
#define HH 8
#define EE 8192
#define EAA 32
#define HDD 32

typedef __attribute__((ext_vector_type(16))) _Float16 v16h;
typedef __attribute__((ext_vector_type(8)))  float    v8f;

// Fragment = 16 f16 = 32 bytes = two uint4 loads.
union Frag { v16h v; uint4 q[2]; _Float16 h[16]; };

#define LOG_CLIP -13.815510557964274f  /* ln(1e-6) */
#define INV_SQRT_HD 0.17677669529663687f

// ---------------- Kernel 0: pack x -> f16 row-major, W -> f16 transposed ----------------
__global__ void __launch_bounds__(256) pack_inputs(const float* __restrict__ x,
                                                   const float* __restrict__ W,
                                                   _Float16* __restrict__ xh,
                                                   _Float16* __restrict__ wth) {
  const int idx = blockIdx.x * blockDim.x + threadIdx.x;   // 0 .. B*N*D-1
  xh[idx] = (_Float16)x[idx];
  if (idx < 3 * DD * DD) {                                 // 768*256 Wt elements
    const int n = idx / DD, k = idx % DD;
    wth[idx] = (_Float16)W[k * (3 * DD) + n];
  }
}

// ---------------- Kernel 1: qkv GEMM (f16 in, f16 out; V written transposed) ----------------
__global__ void __launch_bounds__(256) qkv_gemm(const _Float16* __restrict__ xh,
                                                const _Float16* __restrict__ wth,
                                                const float* __restrict__ bias,
                                                _Float16* __restrict__ qkh,
                                                _Float16* __restrict__ vpack) {
  const int wave = blockIdx.x * (blockDim.x >> 5) + (threadIdx.x >> 5);
  const int lane = threadIdx.x & 31;
  const int NT = (3 * DD) / 16;               // 48
  const int mt = wave / NT, nt = wave % NT;   // wave-uniform
  if (mt >= (BB * NN) / 16) return;
  const int m0 = mt * 16, n0 = nt * 16;
  const int am = m0 + (lane & 15);
  const int bn = n0 + (lane & 15);
  const int kb   = (lane & 16) ? 8 : 0;       // A-frag K sub-base
  const int base = (lane & 16) ? 16 : 0;      // unused; B-frag also uses kb
  const _Float16* arow = xh + (size_t)am * DD;
  const _Float16* brow = wth + (size_t)bn * DD;
  v8f acc = {};
  for (int k0 = 0; k0 < DD; k0 += 32) {
    Frag a, b;
    a.q[0] = *(const uint4*)(arow + k0 + kb);
    a.q[1] = *(const uint4*)(arow + k0 + 16 + kb);
    b.q[0] = *(const uint4*)(brow + k0 + kb);
    b.q[1] = *(const uint4*)(brow + k0 + 16 + kb);
    acc = __builtin_amdgcn_wmma_f32_16x16x32_f16(false, a.v, false, b.v, (short)0, acc, false, false);
  }
  const float bv = bias[bn];
  if (n0 < 2 * DD) {            // Q,K columns -> row-major f16
#pragma unroll
    for (int r = 0; r < 8; ++r) {
      const int m = m0 + r + ((lane & 16) ? 8 : 0);
      qkh[(size_t)m * (3 * DD) + bn] = (_Float16)(acc[r] + bv);
    }
  } else {                      // V columns -> transposed pack [b,h,hd][n]
    const int vc = bn - 2 * DD;
    const int h = vc >> 5, hd = vc & 31;
#pragma unroll
    for (int r = 0; r < 8; ++r) {
      const int m = m0 + r + ((lane & 16) ? 8 : 0);
      const int bb = m >> 9, krow = m & 511;
      vpack[((size_t)((bb * HH + h) * HDD + hd)) * NN + krow] = (_Float16)(acc[r] + bv);
    }
  }
}

// ---------------- Kernel 2: adjusted = Q.Kt/sqrt(hd) + max(z_baseline, ln 1e-6) ----------------
__global__ void __launch_bounds__(256) scores_baseline(const _Float16* __restrict__ qkh,
                                                       const float* __restrict__ adj,
                                                       const float* __restrict__ shifts,
                                                       const float* __restrict__ widths,
                                                       float* __restrict__ adjusted) {
  const int wave = blockIdx.x * (blockDim.x >> 5) + (threadIdx.x >> 5);
  const int lane = threadIdx.x & 31;
  int t = wave;
  const int nt = t & 31;  t >>= 5;
  const int mt = t & 31;  t >>= 5;
  const int h  = t & 7;   t >>= 3;
  const int b  = t;
  if (b >= BB) return;
  const int m0 = mt * 16, n0 = nt * 16;
  const int am = m0 + (lane & 15);
  const int bn = n0 + (lane & 15);
  const int kb   = (lane & 16) ? 8 : 0;
  const int base = (lane & 16) ? 16 : 0;
  const _Float16* Qrow = qkh + (size_t)(b * NN + am) * (3 * DD) + h * HDD;
  const _Float16* Krow = qkh + (size_t)(b * NN + bn) * (3 * DD) + DD + h * HDD;
  Frag a, bb;
  a.q[0]  = *(const uint4*)(Qrow + kb);
  a.q[1]  = *(const uint4*)(Qrow + 16 + kb);
  bb.q[0] = *(const uint4*)(Krow + kb);
  bb.q[1] = *(const uint4*)(Krow + 16 + kb);
  v8f acc = {};
  acc = __builtin_amdgcn_wmma_f32_16x16x32_f16(false, a.v, false, bb.v, (short)0, acc, false, false);
  const float sh = shifts[h], wd = widths[h];
  const float inv2w = 1.0f / (2.0f * wd * wd + 1e-6f);
  float* outp = adjusted + ((size_t)(b * HH + h)) * NN * NN;
#pragma unroll
  for (int r = 0; r < 8; ++r) {
    const int m = m0 + r + ((lane & 16) ? 8 : 0);
    const float s = acc[r] * INV_SQRT_HD;
    const float d = adj[((size_t)(b * NN + m)) * NN + bn];
    const float z = -(d - sh) * (d - sh) * inv2w;
    outp[(size_t)m * NN + bn] = s + fmaxf(z, LOG_CLIP);
  }
}

// ---------------- Kernel 3: edge scatter ----------------
__global__ void __launch_bounds__(256) edge_scatter(const float* __restrict__ adj,
                                                    const int* __restrict__ edge_index,
                                                    const float* __restrict__ edge_attr,
                                                    const unsigned char* __restrict__ valid,
                                                    const float* __restrict__ dsW, const float* __restrict__ dsb,
                                                    const float* __restrict__ dwW, const float* __restrict__ dwb,
                                                    const float* __restrict__ shifts,
                                                    const float* __restrict__ widths,
                                                    float* __restrict__ adjusted) {
  const int idx = blockIdx.x * blockDim.x + threadIdx.x;
  if (idx >= BB * EE * HH) return;
  const int h = idx & 7;
  const int e = (idx >> 3) & (EE - 1);
  const int b = idx >> 16;
  const int eu = edge_index[(b * 2 + 0) * EE + e];
  const int ev = edge_index[(b * 2 + 1) * EE + e];
  const float d = adj[((size_t)(b * NN + eu)) * NN + ev];
  const float* ea = edge_attr + ((size_t)(b * EE + e)) * EAA;
  float ds = dsb[h], dw = dwb[h];
#pragma unroll 8
  for (int k = 0; k < EAA; ++k) {
    const float v = ea[k];
    ds = fmaf(v, dsW[k * HH + h], ds);
    dw = fmaf(v, dwW[k * HH + h], dw);
  }
  const float msh = shifts[h] + ds, mwd = widths[h] + dw;
  const float zm = -(d - msh) * (d - msh) / (2.f * mwd * mwd + 1e-6f);
  const float mlog = valid[b * EE + e] ? fmaxf(zm, LOG_CLIP) : 0.f;
  const float sh = shifts[h], wd = widths[h];
  const float zb = -(d - sh) * (d - sh) / (2.f * wd * wd + 1e-6f);
  const float lb = fmaxf(zb, LOG_CLIP);
  const size_t off = (((size_t)b * HH + h) * NN + eu) * NN + ev;
  adjusted[off] = adjusted[off] - lb + mlog;
}

// ---------------- Kernel 4: softmax + attn @ V (async V tile into LDS) ----------------
__global__ void __launch_bounds__(64) softmax_av(const float* __restrict__ adjusted,
                                                 const _Float16* __restrict__ vpack,
                                                 const unsigned char* __restrict__ mask,
                                                 const float* __restrict__ self_loop,
                                                 float* __restrict__ out) {
  __shared__ _Float16 probs[2][16 * NN];   // 32 KB (16 KB per wave)
  __shared__ _Float16 vtile[HDD * NN];     // 32 KB, shared by both waves (same b,h)
  const int w = threadIdx.x >> 5, lane = threadIdx.x & 31;
  int t = blockIdx.x * 2 + w;
  const int mt = t & 31;  t >>= 5;
  const int h  = t & 7;   t >>= 3;
  const int b  = t;
  const int m0 = mt * 16;

  // ---- issue async copy of V tile (32x512 f16 = 32 KB) into LDS; overlap with softmax ----
  {
    const char* gsrc = (const char*)(vpack + ((size_t)(b * HH + h) * HDD) * NN);
    const unsigned ldsbase = (unsigned)(unsigned long long)&vtile[0];
#pragma unroll
    for (int it = 0; it < 32; ++it) {          // per wave: 32 * 512B = 16 KB
      const int off = w * 16384 + it * 512 + lane * 16;
      const unsigned la = ldsbase + off;
      const char* ga = gsrc + off;
      asm volatile("global_load_async_to_lds_b128 %0, %1, off"
                   :: "v"(la), "v"(ga) : "memory");
    }
  }

  // ---- softmax, one row at a time, 32 lanes x float4 coalesced ----
  const float slw = self_loop[h];
  float rs[16];
#pragma unroll
  for (int r = 0; r < 16; ++r) {
    const int m = m0 + r;
    const float* arow = adjusted + (((size_t)(b * HH + h)) * NN + m) * NN;
    const bool mrow = mask[b * NN + m] != 0;
    float v[16];
#pragma unroll
    for (int j = 0; j < 4; ++j) {
      const int c = lane * 4 + j * 128;
      const float4 f = *(const float4*)(arow + c);
      const unsigned mk = *(const unsigned*)(mask + b * NN + c);
      float vv[4] = {f.x, f.y, f.z, f.w};
#pragma unroll
      for (int q = 0; q < 4; ++q) {
        float val = vv[q] + ((c + q) == m ? slw : 0.f);
        if (!(mrow && ((mk >> (8 * q)) & 0xFF))) val = -1.0e6f;
        v[j * 4 + q] = val;
      }
    }
    float mx = -3.0e38f;
#pragma unroll
    for (int i = 0; i < 16; ++i) mx = fmaxf(mx, v[i]);
    for (int s = 16; s > 0; s >>= 1) mx = fmaxf(mx, __shfl_xor(mx, s, 32));
    float sum = 0.f;
#pragma unroll
    for (int j = 0; j < 4; ++j) {
      const int c = lane * 4 + j * 128;
      union { unsigned long long u; _Float16 h4[4]; } pk;
#pragma unroll
      for (int q = 0; q < 4; ++q) {
        const float p = __expf(v[j * 4 + q] - mx);
        sum += p;
        pk.h4[q] = (_Float16)p;
      }
      *(unsigned long long*)&probs[w][r * NN + c] = pk.u;   // ds_store_b64
    }
    for (int s = 16; s > 0; s >>= 1) sum += __shfl_xor(sum, s, 32);
    rs[r] = sum;
  }

  asm volatile("s_wait_asynccnt 0x0" ::: "memory");
  __syncthreads();

  // ---- attn @ V: M=16, N=32 (two accs), K=512; all fragments from LDS via b128 ----
  v8f acc0 = {}, acc1 = {};
  const int an   = lane & 15;
  const int kb   = (lane & 16) ? 8 : 0;
  const int base = (lane & 16) ? 16 : 0;
  const _Float16* prow = &probs[w][(lane & 15) * NN];
  for (int kc = 0; kc < NN; kc += 32) {
    Frag a, b0, b1;
    a.q[0]  = *(const uint4*)(prow + kc + kb);
    a.q[1]  = *(const uint4*)(prow + kc + 16 + kb);
    b0.q[0] = *(const uint4*)(&vtile[an * NN + kc + kb]);
    b0.q[1] = *(const uint4*)(&vtile[an * NN + kc + 16 + kb]);
    b1.q[0] = *(const uint4*)(&vtile[(16 + an) * NN + kc + kb]);
    b1.q[1] = *(const uint4*)(&vtile[(16 + an) * NN + kc + 16 + kb]);
    acc0 = __builtin_amdgcn_wmma_f32_16x16x32_f16(false, a.v, false, b0.v, (short)0, acc0, false, false);
    acc1 = __builtin_amdgcn_wmma_f32_16x16x32_f16(false, a.v, false, b1.v, (short)0, acc1, false, false);
  }
#pragma unroll
  for (int rr = 0; rr < 8; ++rr) {
    const float inv = 1.0f / ((lane & 16) ? rs[rr + 8] : rs[rr]);
    const int gm = m0 + rr + ((lane & 16) ? 8 : 0);
    float* orow = out + ((size_t)(b * NN + gm)) * DD + h * HDD;
    orow[an]      = acc0[rr] * inv;
    orow[16 + an] = acc1[rr] * inv;
  }
}

extern "C" void kernel_launch(void* const* d_in, const int* in_sizes, int n_in,
                              void* d_out, int out_size, void* d_ws, size_t ws_size,
                              hipStream_t stream) {
  const float* x          = (const float*)d_in[0];
  const float* adj        = (const float*)d_in[1];
  const int*   edge_index = (const int*)d_in[2];
  const float* edge_attr  = (const float*)d_in[3];
  const unsigned char* mask  = (const unsigned char*)d_in[4];
  const unsigned char* vmask = (const unsigned char*)d_in[5];
  const float* qkv_W = (const float*)d_in[6];
  const float* qkv_b = (const float*)d_in[7];
  const float* ds_W  = (const float*)d_in[12];
  const float* ds_b  = (const float*)d_in[13];
  const float* dw_W  = (const float*)d_in[14];
  const float* dw_b  = (const float*)d_in[15];
  const float* shifts = (const float*)d_in[16];
  const float* widths = (const float*)d_in[17];
  const float* slw    = (const float*)d_in[18];
  float* out = (float*)d_out;

  char* ws = (char*)d_ws;
  size_t off = 0;
  _Float16* xh   = (_Float16*)(ws + off); off += (size_t)BB * NN * DD * 2;        // 2 MB
  _Float16* wth  = (_Float16*)(ws + off); off += (size_t)3 * DD * DD * 2;         // 384 KB
  _Float16* qkh  = (_Float16*)(ws + off); off += (size_t)BB * NN * 3 * DD * 2;    // 6 MB
  _Float16* vpk  = (_Float16*)(ws + off); off += (size_t)BB * HH * HDD * NN * 2;  // 2 MB
  float* adjusted = (float*)(ws + off);                                           // 64 MB

  pack_inputs<<<(BB * NN * DD) / 256, 256, 0, stream>>>(x, qkv_W, xh, wth);
  qkv_gemm<<<1536, 256, 0, stream>>>(xh, wth, qkv_b, qkh, vpk);
  scores_baseline<<<8192, 256, 0, stream>>>(qkh, adj, shifts, widths, adjusted);
  edge_scatter<<<(BB * EE * HH) / 256, 256, 0, stream>>>(adj, edge_index, edge_attr, vmask,
                                                         ds_W, ds_b, dw_W, dw_b,
                                                         shifts, widths, adjusted);
  softmax_av<<<1024, 64, 0, stream>>>(adjusted, vpk, mask, slw, out);
}
